// AMKBlock_81750407512188
// MI455X (gfx1250) — matrix operand order repaired and checked by reading, not compiled, not run on hardware
//
#include <hip/hip_runtime.h>

// ---------------------------------------------------------------------------
// Transformer block for MI455X (gfx1250, wave32, WMMA 16x16x32 f16).
// ~140 GFLOP vs ~150MB traffic -> matrix-core bound. GEMMs: double-buffered
// async global->LDS staging (ASYNCcnt) feeding v_wmma_f32_16x16x32_f16.
// ---------------------------------------------------------------------------

typedef _Float16 half_t;
typedef __attribute__((ext_vector_type(16))) _Float16 v16h;
typedef __attribute__((ext_vector_type(8)))  _Float16 v8h;
typedef __attribute__((ext_vector_type(8)))  float    v8f;

#define Bb    2
#define Nn    2048
#define Dd    1024
#define Hh    16
#define DHd   64
#define INTERi 2816
#define Mm    (Bb * Nn)          // 4096 token rows
#define EPSf  1e-5f

// ---------------------------------------------------------------------------
// WMMA helpers (fragment layouts per CDNA5 ISA §7.12.2)
// ---------------------------------------------------------------------------
static __device__ __forceinline__ v8f wmma32(v16h a, v16h b, v8f c) {
  return __builtin_amdgcn_wmma_f32_16x16x32_f16(
      /*neg_a=*/false, a, /*neg_b=*/false, b,
      /*c_mod=*/(short)0, c, /*reuse_a=*/false, /*reuse_b=*/false);
}

// A-matrix 16x32 f16: lane L (row = L&15): VGPR0-3 hold K = (L>>4)*8 .. +7,
// VGPR4-7 hold K = 16+(L>>4)*8 .. +7.  Two 16B row chunks per lane.
static __device__ __forceinline__ v16h load_a_frag(const half_t* p0, int ld) {
  const int lane = threadIdx.x & 31;
  const int r = lane & 15, hi = lane >> 4;
  const half_t* p = p0 + (size_t)r * ld + hi * 8;
  v8h lo = *(const v8h*)(p);
  v8h hq = *(const v8h*)(p + 16);
  v16h o;
#pragma unroll
  for (int i = 0; i < 8; ++i) { o[i] = lo[i]; o[i + 8] = hq[i]; }
  return o;
}

// B-matrix 32x16 f16 (B[k][n] = row-major source row n, K contiguous):
// lane L holds column n = L&15, K = (L>>4)*8 .. +7 and 16+(L>>4)*8 .. +7.
static __device__ __forceinline__ v16h load_b_frag(const half_t* p0, int ld) {
  const int lane = threadIdx.x & 31;
  const int c = lane & 15, hi = lane >> 4;
  const half_t* p = p0 + (size_t)c * ld + hi * 8;
  v8h lo = *(const v8h*)(p);
  v8h hq = *(const v8h*)(p + 16);
  v16h o;
#pragma unroll
  for (int i = 0; i < 8; ++i) { o[i] = lo[i]; o[i + 8] = hq[i]; }
  return o;
}

// Async DMA of one 16-byte chunk global -> LDS (ASYNCcnt-tracked).
// VDST = VGPR holding LDS byte address, VADDR = 64-bit global address (GV mode).
static __device__ __forceinline__ void async_copy16(const half_t* g, half_t* l) {
  unsigned lds_addr = (unsigned)(unsigned long long)(void*)l;
  asm volatile("global_load_async_to_lds_b128 %0, %1, off"
               :: "v"(lds_addr), "v"(g) : "memory");
}

// ---------------------------------------------------------------------------
// fp32 -> f16 convert
// ---------------------------------------------------------------------------
__global__ __launch_bounds__(256) void cvt_f32_f16(const float* __restrict__ x,
                                                   half_t* __restrict__ y, int n) {
  int i = blockIdx.x * 256 + threadIdx.x;
  if (i < n) y[i] = (half_t)x[i];
}

// ---------------------------------------------------------------------------
// Generic GEMM: C[m,n] = sum_k A[m,k] * W[n,k]   (A: MxK f16, W: NxK f16)
// 256 threads = 8 waves; block tile 128x128, wave tile 32x64 (2x4 WMMA tiles).
// Double-buffered async global->LDS staging of 128x32 A and W slabs.
// ---------------------------------------------------------------------------
__global__ __launch_bounds__(256)
void gemm_f16_wmma(const half_t* __restrict__ A, const half_t* __restrict__ W,
                   float* __restrict__ C, int M, int N, int K) {
  __shared__ half_t ldsA[2][128 * 32];   // 8KB per buffer
  __shared__ half_t ldsB[2][128 * 32];   // 8KB per buffer
  const int t = threadIdx.x;
  const int w = t >> 5, wm = w & 3, wn = w >> 2;
  const int bm = blockIdx.x * 128, bn = blockIdx.y * 128;
  (void)M;

  // Each thread DMAs 2x16B of A and 2x16B of W per K-slab (4 async ops).
  auto issue = [&](int kk, int buf) {
#pragma unroll
    for (int i = 0; i < 2; ++i) {
      int c = t * 2 + i;               // 0..511 16B-chunks per slab
      int row = c >> 2, col = (c & 3) * 8;
      async_copy16(A + (size_t)(bm + row) * K + kk + col, &ldsA[buf][row * 32 + col]);
      async_copy16(W + (size_t)(bn + row) * K + kk + col, &ldsB[buf][row * 32 + col]);
    }
  };

  v8f acc[2][4];
#pragma unroll
  for (int i = 0; i < 2; ++i)
#pragma unroll
    for (int j = 0; j < 4; ++j) acc[i][j] = (v8f){0.f,0.f,0.f,0.f,0.f,0.f,0.f,0.f};

  const int T = K / 32;
  issue(0, 0);
  for (int i = 0; i < T; ++i) {
    const int buf = i & 1;
    if (i + 1 < T) {
      issue((i + 1) * 32, buf ^ 1);                      // next slab in flight
      asm volatile("s_wait_asynccnt 4" ::: "memory");    // oldest batch (4) done
    } else {
      asm volatile("s_wait_asynccnt 0" ::: "memory");
    }
    __syncthreads();                                     // slab i visible to all waves
    v16h a0 = load_a_frag(&ldsA[buf][(wm * 32) * 32], 32);
    v16h a1 = load_a_frag(&ldsA[buf][(wm * 32 + 16) * 32], 32);
#pragma unroll
    for (int j = 0; j < 4; ++j) {
      v16h b = load_b_frag(&ldsB[buf][(wn * 64 + 16 * j) * 32], 32);
      acc[0][j] = wmma32(a0, b, acc[0][j]);
      acc[1][j] = wmma32(a1, b, acc[1][j]);
    }
    __syncthreads();                                     // done reading before overwrite
  }

  // C/D layout: VGPR r, lanes 0-15 -> row r, lanes 16-31 -> row 8+r; col = lane&15
  const int lane = threadIdx.x & 31;
  const int cl = lane & 15, hi = lane >> 4;
  const int m0 = bm + wm * 32, n0 = bn + wn * 64;
#pragma unroll
  for (int i = 0; i < 2; ++i)
#pragma unroll
    for (int j = 0; j < 4; ++j)
#pragma unroll
      for (int r = 0; r < 8; ++r) {
        int row = m0 + 16 * i + r + 8 * hi;
        int col = n0 + 16 * j + cl;
        C[(size_t)row * N + col] = acc[i][j][r];
      }
}

// ---------------------------------------------------------------------------
// RoPE + per-head RMS on q,k; repack q,k -> [b,h,n,dh] f16 and v -> [b,h,dh,n]
// One wave per (b,n,h); lane handles dh = lane and lane+32.
// ---------------------------------------------------------------------------
__global__ __launch_bounds__(128)
void rope_rms_pack(const float* __restrict__ qkv, const float* __restrict__ cosT,
                   const float* __restrict__ sinT, half_t* __restrict__ qh,
                   half_t* __restrict__ kh, half_t* __restrict__ vT) {
  const int gw = blockIdx.x * 4 + (threadIdx.x >> 5);
  const int lane = threadIdx.x & 31;
  const int h = gw % Hh;
  const int n = (gw / Hh) % Nn;
  const int b = gw / (Hh * Nn);
  const float* base = qkv + ((size_t)(b * Nn + n)) * 3 * Dd + h * DHd;
  const float c0 = cosT[n * DHd + lane], c1 = cosT[n * DHd + lane + 32];
  const float s0 = sinT[n * DHd + lane], s1 = sinT[n * DHd + lane + 32];

#pragma unroll
  for (int s = 0; s < 2; ++s) {  // 0 = q, 1 = k
    float x0 = base[s * Dd + lane];
    float x1 = base[s * Dd + lane + 32];
    float r0 = x0 * c0 - x1 * s0;   // rotate_half: [-x2, x1]
    float r1 = x1 * c1 + x0 * s1;
    float ss = r0 * r0 + r1 * r1;
#pragma unroll
    for (int msk = 16; msk >= 1; msk >>= 1) ss += __shfl_xor(ss, msk, 32);
    float rn = rsqrtf(ss * (1.0f / DHd) + EPSf);
    half_t* dst = (s == 0 ? qh : kh) + (((size_t)(b * Hh + h)) * Nn + n) * DHd;
    dst[lane]      = (half_t)(r0 * rn);
    dst[lane + 32] = (half_t)(r1 * rn);
  }
  // v transposed: vT[b,h,dh,n] so PV B-fragments are contiguous-K row loads
  half_t* vt = vT + ((size_t)(b * Hh + h)) * DHd * Nn;
  vt[(size_t)lane * Nn + n]        = (half_t)base[2 * Dd + lane];
  vt[(size_t)(lane + 32) * Nn + n] = (half_t)base[2 * Dd + lane + 32];
}

// ---------------------------------------------------------------------------
// Flash attention: one wave per (b,h, 16-row q tile). Online softmax with
// half-wave shuffles matching the C/D fragment layout; P staged through LDS
// and re-read as an A-fragment (ds_store -> s_wait_dscnt -> ds_load).
// Epilogue fuses m = cvec - v, stores f16 [b,n,d].
// ---------------------------------------------------------------------------
__global__ __launch_bounds__(128)
void attn_flash(const half_t* __restrict__ qh, const half_t* __restrict__ kh,
                const half_t* __restrict__ vT, const float* __restrict__ qkv,
                half_t* __restrict__ mOut) {
  __shared__ half_t ldsP[4][16 * 40];  // per-wave 16x32 P tile, stride 40 halves
  const int wv = threadIdx.x >> 5;
  const int lane = threadIdx.x & 31;
  const int cl = lane & 15, hi = lane >> 4;
  const int gw = blockIdx.x * 4 + wv;
  const int qt = gw % (Nn / 16);
  const int bh = gw / (Nn / 16);
  const int h = bh % Hh, b = bh / Hh;
  const int q0 = qt * 16;

  const half_t* Q  = qh + ((size_t)bh * Nn + q0) * DHd;
  const half_t* Kp = kh + (size_t)bh * Nn * DHd;
  const half_t* Vt = vT + (size_t)bh * DHd * Nn;

  v16h aq0 = load_a_frag(Q, DHd);        // dh 0..31
  v16h aq1 = load_a_frag(Q + 32, DHd);   // dh 32..63

  v8f o[4];
#pragma unroll
  for (int t = 0; t < 4; ++t) o[t] = (v8f){0.f,0.f,0.f,0.f,0.f,0.f,0.f,0.f};
  float mrow[8], lrow[8];
#pragma unroll
  for (int r = 0; r < 8; ++r) { mrow[r] = -1e30f; lrow[r] = 0.f; }

  for (int kb = 0; kb < Nn; kb += 32) {
    v8f s[2];
#pragma unroll
    for (int j = 0; j < 2; ++j) {
      const half_t* kp = Kp + (size_t)(kb + 16 * j) * DHd;
      v8f t = (v8f){0.f,0.f,0.f,0.f,0.f,0.f,0.f,0.f};
      t = wmma32(aq0, load_b_frag(kp, DHd), t);
      t = wmma32(aq1, load_b_frag(kp + 32, DHd), t);
#pragma unroll
      for (int r = 0; r < 8; ++r) t[r] *= 0.125f;  // 1/sqrt(64)
      s[j] = t;
    }
    float f[8];
#pragma unroll
    for (int r = 0; r < 8; ++r) {
      float mx = fmaxf(s[0][r], s[1][r]);
#pragma unroll
      for (int msk = 8; msk >= 1; msk >>= 1) mx = fmaxf(mx, __shfl_xor(mx, msk, 32));
      float mn = fmaxf(mrow[r], mx);
      f[r] = __expf(mrow[r] - mn);
      mrow[r] = mn;
      lrow[r] *= f[r];
    }
#pragma unroll
    for (int t = 0; t < 4; ++t)
#pragma unroll
      for (int r = 0; r < 8; ++r) o[t][r] *= f[r];
#pragma unroll
    for (int j = 0; j < 2; ++j)
#pragma unroll
      for (int r = 0; r < 8; ++r) {
        float p = __expf(s[j][r] - mrow[r]);
        float ps = p;
#pragma unroll
        for (int msk = 8; msk >= 1; msk >>= 1) ps += __shfl_xor(ps, msk, 32);
        lrow[r] += ps;
        ldsP[wv][(r + 8 * hi) * 40 + 16 * j + cl] = (half_t)p;
      }
    asm volatile("s_wait_dscnt 0" ::: "memory");  // order ds_store -> ds_load (same wave)
    v16h ap = load_a_frag(&ldsP[wv][0], 40);
#pragma unroll
    for (int t = 0; t < 4; ++t)
      o[t] = wmma32(ap, load_b_frag(Vt + (size_t)(16 * t) * Nn + kb, Nn), o[t]);
  }

#pragma unroll
  for (int t = 0; t < 4; ++t)
#pragma unroll
    for (int r = 0; r < 8; ++r) {
      int row = q0 + r + 8 * hi;
      int dh = 16 * t + cl;
      float val = o[t][r] / lrow[r];
      float vv = qkv[((size_t)(b * Nn + row)) * 3 * Dd + 2 * Dd + h * DHd + dh];
      mOut[((size_t)(b * Nn + row)) * Dd + h * DHd + dh] = (half_t)(val - vv);
    }
}

// ---------------------------------------------------------------------------
// attn -= dot(attn,hs)*hs; h = rms_norm(hs + attn); store f32 + f16
// ---------------------------------------------------------------------------
__global__ __launch_bounds__(256)
void resid_proj_rms(const float* __restrict__ attn, const float* __restrict__ hs,
                    float* __restrict__ hF, half_t* __restrict__ hH) {
  __shared__ float red[8];
  const int m = blockIdx.x, t = threadIdx.x;
  const int lane = t & 31, wv = t >> 5;
  const float* a = attn + (size_t)m * Dd;
  const float* x = hs + (size_t)m * Dd;
  float av[4], xv[4], dot = 0.f;
#pragma unroll
  for (int i = 0; i < 4; ++i) {
    int idx = t + 256 * i;
    av[i] = a[idx]; xv[i] = x[idx]; dot += av[i] * xv[i];
  }
#pragma unroll
  for (int msk = 16; msk >= 1; msk >>= 1) dot += __shfl_xor(dot, msk, 32);
  if (lane == 0) red[wv] = dot;
  __syncthreads();
  float d = 0.f;
#pragma unroll
  for (int w = 0; w < 8; ++w) d += red[w];
  __syncthreads();
  float tv[4], ss = 0.f;
#pragma unroll
  for (int i = 0; i < 4; ++i) { tv[i] = xv[i] + av[i] - d * xv[i]; ss += tv[i] * tv[i]; }
#pragma unroll
  for (int msk = 16; msk >= 1; msk >>= 1) ss += __shfl_xor(ss, msk, 32);
  if (lane == 0) red[wv] = ss;
  __syncthreads();
  float sst = 0.f;
#pragma unroll
  for (int w = 0; w < 8; ++w) sst += red[w];
  float rn = rsqrtf(sst * (1.0f / Dd) + EPSf);
#pragma unroll
  for (int i = 0; i < 4; ++i) {
    int idx = t + 256 * i;
    float hv = tv[i] * rn;
    hF[(size_t)m * Dd + idx] = hv;
    hH[(size_t)m * Dd + idx] = (half_t)hv;
  }
}

// act = silu(gate) * up, f16 output for the down GEMM
__global__ __launch_bounds__(256)
void silu_mul(const float* __restrict__ gu, half_t* __restrict__ act, int n) {
  int i = blockIdx.x * 256 + threadIdx.x;
  if (i >= n) return;
  int m = i / INTERi, c = i % INTERi;
  float g = gu[(size_t)m * (2 * INTERi) + c];
  float u = gu[(size_t)m * (2 * INTERi) + INTERi + c];
  float sg = g / (1.f + __expf(-g));
  act[i] = (half_t)(sg * u);
}

// out = rms_norm(h + mlp)
__global__ __launch_bounds__(256)
void final_rms(const float* __restrict__ hF, const float* __restrict__ mlp,
               float* __restrict__ out) {
  __shared__ float red[8];
  const int m = blockIdx.x, t = threadIdx.x;
  const int lane = t & 31, wv = t >> 5;
  float tv[4], ss = 0.f;
#pragma unroll
  for (int i = 0; i < 4; ++i) {
    int idx = t + 256 * i;
    tv[i] = hF[(size_t)m * Dd + idx] + mlp[(size_t)m * Dd + idx];
    ss += tv[i] * tv[i];
  }
#pragma unroll
  for (int msk = 16; msk >= 1; msk >>= 1) ss += __shfl_xor(ss, msk, 32);
  if (lane == 0) red[wv] = ss;
  __syncthreads();
  float sst = 0.f;
#pragma unroll
  for (int w = 0; w < 8; ++w) sst += red[w];
  float rn = rsqrtf(sst * (1.0f / Dd) + EPSf);
#pragma unroll
  for (int i = 0; i < 4; ++i) {
    int idx = t + 256 * i;
    out[(size_t)m * Dd + idx] = tv[i] * rn;
  }
}

// ---------------------------------------------------------------------------
extern "C" void kernel_launch(void* const* d_in, const int* in_sizes, int n_in,
                              void* d_out, int out_size, void* d_ws, size_t ws_size,
                              hipStream_t stream) {
  (void)in_sizes; (void)n_in; (void)out_size; (void)ws_size;
  const float* hs    = (const float*)d_in[0];
  const float* cosT  = (const float*)d_in[1];
  const float* sinT  = (const float*)d_in[2];
  const float* w_qkv = (const float*)d_in[3];
  const float* w_o   = (const float*)d_in[4];
  const float* w_gu  = (const float*)d_in[5];
  const float* w_dn  = (const float*)d_in[6];
  float* out = (float*)d_out;

  char* ws = (char*)d_ws;
  size_t off = 0;
  auto alloc = [&](size_t bytes) -> void* {
    void* p = ws + off;
    off += (bytes + 255) & ~(size_t)255;
    return p;
  };
  half_t* hs_h   = (half_t*)alloc((size_t)Mm * Dd * 2);
  half_t* wqkv_h = (half_t*)alloc((size_t)3 * Dd * Dd * 2);
  half_t* wo_h   = (half_t*)alloc((size_t)Dd * Dd * 2);
  half_t* wgu_h  = (half_t*)alloc((size_t)2 * INTERi * Dd * 2);
  half_t* wdn_h  = (half_t*)alloc((size_t)Dd * INTERi * 2);
  float*  qkv_f  = (float*)alloc((size_t)Mm * 3 * Dd * 4);
  half_t* q_h    = (half_t*)alloc((size_t)Bb * Hh * Nn * DHd * 2);
  half_t* k_h    = (half_t*)alloc((size_t)Bb * Hh * Nn * DHd * 2);
  half_t* vT_h   = (half_t*)alloc((size_t)Bb * Hh * DHd * Nn * 2);
  half_t* m_h    = (half_t*)alloc((size_t)Mm * Dd * 2);
  float*  attn_f = (float*)alloc((size_t)Mm * Dd * 4);
  float*  h_f    = (float*)alloc((size_t)Mm * Dd * 4);
  half_t* h_h    = (half_t*)alloc((size_t)Mm * Dd * 2);
  float*  gu_f   = (float*)alloc((size_t)Mm * 2 * INTERi * 4);
  half_t* act_h  = (half_t*)alloc((size_t)Mm * INTERi * 2);
  float*  mlp_f  = (float*)alloc((size_t)Mm * Dd * 4);

  auto cvt = [&](const float* src, half_t* dst, int n) {
    cvt_f32_f16<<<(n + 255) / 256, 256, 0, stream>>>(src, dst, n);
  };
  cvt(hs,    hs_h,   Mm * Dd);
  cvt(w_qkv, wqkv_h, 3 * Dd * Dd);
  cvt(w_o,   wo_h,   Dd * Dd);
  cvt(w_gu,  wgu_h,  2 * INTERi * Dd);
  cvt(w_dn,  wdn_h,  Dd * INTERi);

  // QKV projection: 4096 x 3072 x 1024
  gemm_f16_wmma<<<dim3(Mm / 128, (3 * Dd) / 128), 256, 0, stream>>>(
      hs_h, wqkv_h, qkv_f, Mm, 3 * Dd, Dd);

  // RoPE + q/k RMS + repack (one wave per (b,n,h))
  rope_rms_pack<<<(Bb * Nn * Hh) / 4, 128, 0, stream>>>(qkv_f, cosT, sinT, q_h, k_h, vT_h);

  // Flash attention + (cvec - v) epilogue (one wave per 16 q rows)
  attn_flash<<<(Bb * Hh * (Nn / 16)) / 4, 128, 0, stream>>>(q_h, k_h, vT_h, qkv_f, m_h);

  // Output projection: 4096 x 1024 x 1024
  gemm_f16_wmma<<<dim3(Mm / 128, Dd / 128), 256, 0, stream>>>(
      m_h, wo_h, attn_f, Mm, Dd, Dd);

  // attn -= dot*hs; h = rms(hs + attn)
  resid_proj_rms<<<Mm, 256, 0, stream>>>(attn_f, hs, h_f, h_h);

  // gate_up: 4096 x 5632 x 1024
  gemm_f16_wmma<<<dim3(Mm / 128, (2 * INTERi) / 128), 256, 0, stream>>>(
      h_h, wgu_h, gu_f, Mm, 2 * INTERi, Dd);

  // act = silu(gate) * up
  silu_mul<<<(Mm * INTERi + 255) / 256, 256, 0, stream>>>(gu_f, act_h, Mm * INTERi);

  // down: 4096 x 1024 x 2816
  gemm_f16_wmma<<<dim3(Mm / 128, Dd / 128), 256, 0, stream>>>(
      act_h, wdn_h, mlp_f, Mm, Dd, INTERi);

  // out = rms(h + mlp)
  final_rms<<<Mm, 256, 0, stream>>>(h_f, mlp_f, out);
}
